// GNN_9534827397531
// MI455X (gfx1250) — compile-verified, matrix operand order checked by reading, not run on hardware
//
#include <hip/hip_runtime.h>
#include <hip/hip_bf16.h>
#include <cstdint>

typedef __bf16 bf16_t;
typedef __bf16 v16bf __attribute__((ext_vector_type(16)));
typedef float  v8f   __attribute__((ext_vector_type(8)));

#define N_NODES 2000
#define C_IN    64
#define C       256
#define BATCH   128
#define NEDGE   64000
#define C_EMB   128
#define HAZ     512
#define INP_W   (C_IN + N_NODES)   // 2064

#if defined(__has_builtin)
#  if __has_builtin(__builtin_amdgcn_global_load_async_to_lds_b128)
#    define HAVE_ASYNC_LDS 1
#  endif
#endif

#ifdef HAVE_ASYNC_LDS
// Builtin signature (from hipcc diagnostic): param0 is
// '__attribute__((__vector_size__(16))) int __device__ *'  (AS1, GCC-vector int4)
typedef int v4i_g __attribute__((vector_size(16)));
typedef __attribute__((address_space(1))) v4i_g* gas_v4i;
typedef __attribute__((address_space(3))) v4i_g* las_v4i;
#endif

// ---------------------------------------------------------------- utilities
__global__ void k_zero_f32(float* p, int n) {
    int i = blockIdx.x * blockDim.x + threadIdx.x;
    if (i < n) p[i] = 0.0f;
}
__global__ void k_zero_i32(int* p, int n) {
    int i = blockIdx.x * blockDim.x + threadIdx.x;
    if (i < n) p[i] = 0;
}

// ------------------------------------------------------- degrees + CSR scan
__global__ void k_count(const int* __restrict__ src, const int* __restrict__ dst,
                        int* deg_src, int* deg_dst) {
    int e = blockIdx.x * blockDim.x + threadIdx.x;
    if (e < NEDGE) {
        atomicAdd(&deg_src[src[e]], 1);
        atomicAdd(&deg_dst[dst[e]], 1);
    }
}

// thread 0 serially builds row_start (2001 entries); all threads compute rsqrt scales
__global__ void k_scan(const int* __restrict__ deg_src, const int* __restrict__ deg_dst,
                       float* dout_is, float* din_is, int* row_start) {
    if (threadIdx.x == 0) {
        int acc = 0;
        row_start[0] = 0;
        for (int n = 0; n < N_NODES; ++n) {
            acc += deg_dst[n];
            row_start[n + 1] = acc;
        }
    }
    for (int n = threadIdx.x; n < N_NODES; n += blockDim.x) {
        int ds = deg_src[n]; if (ds < 1) ds = 1;
        int dd = deg_dst[n]; if (dd < 1) dd = 1;
        dout_is[n] = rsqrtf((float)ds);
        din_is[n]  = rsqrtf((float)dd);
    }
}

// deterministic CSR bucket fill: one thread per dst node, preserves edge order
__global__ void k_bucket(const int* __restrict__ src, const int* __restrict__ dst,
                         const int* __restrict__ row_start, int* sorted_src) {
    int n = blockIdx.x * blockDim.x + threadIdx.x;
    if (n >= N_NODES) return;
    int pos = row_start[n];
    for (int e = 0; e < NEDGE; ++e)
        if (dst[e] == n) sorted_src[pos++] = src[e];
}

// --------------------------------------------- weight prep: f32 -> bf16, transposed
// Wt[n][k] = W[k][n]  (so a WMMA B-fragment column is a contiguous K-run)
__global__ void k_prepW(const float* __restrict__ W, bf16_t* __restrict__ Wt) {
    int n = blockIdx.x;     // 256
    int k = threadIdx.x;    // 256
    Wt[(long)n * C + k] = (bf16_t)W[(long)k * C + n];
}

// ------------------------------------------------------------------- embed
__global__ void k_embed(const float* __restrict__ inp, const float* __restrict__ We,
                        float* __restrict__ emb) {
    int b = blockIdx.x, j = threadIdx.x;          // 128 x 128
    float acc = 0.0f;
    const float* row = inp + (long)b * INP_W;
    for (int k = 0; k < C_IN; ++k) acc += row[k] * We[k * C_EMB + j];
    emb[b * C_EMB + j] = acc > 0.0f ? acc : 0.0f;
}

// ------------------------------------------------ layer 0 (feature dim 1 -> 256)
__global__ __launch_bounds__(256)
void k_layer0(const float* __restrict__ inp, const int* __restrict__ row_start,
              const int* __restrict__ sorted_src, const float* __restrict__ dout_is,
              const float* __restrict__ din_is, const float* __restrict__ W0,
              const float* __restrict__ b0, bf16_t* __restrict__ x1) {
    __shared__ float agg[16];
    const int b  = blockIdx.y;
    const int n0 = blockIdx.x * 16;
    const float* x0 = inp + (long)b * INP_W + C_IN;   // node scalar features
    if (threadIdx.x < 16) {
        int n = n0 + threadIdx.x;
        float acc = 0.0f;
        int s0 = row_start[n], s1 = row_start[n + 1];
        for (int e = s0; e < s1; ++e) {
            int s = sorted_src[e];
            acc += dout_is[s] * x0[s];
        }
        agg[threadIdx.x] = acc * din_is[n];
    }
    __syncthreads();
    const int c  = threadIdx.x;                       // 256 channels
    const float wc = W0[c], bc = b0[c];
    bf16_t* yrow = x1 + ((long)b * N_NODES + n0) * C + c;
    #pragma unroll 4
    for (int i = 0; i < 16; ++i) {
        float v = agg[i] * wc + bc;
        yrow[(long)i * C] = (bf16_t)(v > 0.0f ? v : 0.0f);
    }
}

// ------------------------------------------- CSR gather-aggregate (layers 1,2)
// agg[b][n][c] = din_is[n] * sum_{e in row n} dout_is[src_e] * x[b][src_e][c]
__global__ __launch_bounds__(256)
void k_aggregate(const bf16_t* __restrict__ x, const int* __restrict__ row_start,
                 const int* __restrict__ sorted_src, const float* __restrict__ dout_is,
                 const float* __restrict__ din_is, bf16_t* __restrict__ agg) {
    const int b = blockIdx.y;
    const int c = threadIdx.x;                        // 256 channels
    const bf16_t* xb = x + (long)b * N_NODES * C;
    #pragma unroll 1
    for (int i = 0; i < 8; ++i) {
        int n = blockIdx.x * 8 + i;
        float acc = 0.0f;
        int s0 = row_start[n], s1 = row_start[n + 1];
        for (int e = s0; e < s1; ++e) {
            int s = sorted_src[e];
            float w = dout_is[s];
            acc += w * (float)xb[(long)s * C + c];
        }
        agg[((long)b * N_NODES + n) * C + c] = (bf16_t)(acc * din_is[n]);
    }
}

// ---------------------------------------------------- WMMA bf16 GEMM + bias + ReLU
// Y[m][n] = relu( sum_k A[m][k] * Wt[n][k] + bias[n] ), M = 256000, K = 256, N = 256.
// Block: 256 threads = 8 waves, block tile 64(M) x 128(N); waves arranged 4M x 2N,
// each wave owns a 16x64 strip (four 16x16 D tiles -> one A fragment feeds 4 WMMAs).
// The 128x256 bf16 weight panel (64 KB) is staged in LDS (async global->LDS when
// available), so B fragments are ds_load_b128 and B-side HBM traffic is one panel
// per block. Fragment layout per CDNA5 ISA 7.12.2 (16-bit A): lane L holds row L%16
// with K in {kb..kb+7} U {16+kb..16+kb+7}, kb = (L/16)*8 -> two 16B contiguous loads.
__global__ __launch_bounds__(256)
void k_gemm_bf16(const bf16_t* __restrict__ A, const bf16_t* __restrict__ Bt,
                 const float* __restrict__ bias, bf16_t* __restrict__ Y) {
    __shared__ bf16_t sB[128 * C];                    // 64 KB weight panel

    // ---- stage weight panel: Bt rows [blockIdx.y*128, +128), all K
    const bf16_t* wpan = Bt + (size_t)blockIdx.y * 128 * C;
#ifdef HAVE_ASYNC_LDS
    #pragma unroll
    for (int i = 0; i < 16; ++i) {
        int cid = i * 256 + threadIdx.x;              // 4096 x 16B chunks
        __builtin_amdgcn_global_load_async_to_lds_b128(
            (gas_v4i)(uintptr_t)((const char*)wpan + (size_t)cid * 16),
            (las_v4i)(uintptr_t)(size_t)((char*)sB + (size_t)cid * 16 - (char*)nullptr) + 0,
            0, 0);
    }
    asm volatile("s_wait_asynccnt 0x0" ::: "memory");
#else
    #pragma unroll
    for (int i = 0; i < 16; ++i) {
        int cid = i * 256 + threadIdx.x;
        ((uint4*)sB)[cid] = ((const uint4*)wpan)[cid];
    }
#endif
    __syncthreads();

    const int wave = threadIdx.x >> 5;
    const int lane = threadIdx.x & 31;
    const int mw = wave & 3;                          // 4 M-subtiles
    const int nw = wave >> 2;                         // 2 N-halves
    const int m0 = blockIdx.x * 64 + mw * 16;
    const int l16 = lane & 15;
    const int hi  = lane >> 4;
    const int kb  = hi * 8;

    const bf16_t* arow = A + (long)(m0 + l16) * C + kb;
    const bf16_t* b0l  = sB + (nw * 64 + l16) * C + kb;   // j-th tile: + j*16*C

    v8f acc0 = {0.f,0.f,0.f,0.f,0.f,0.f,0.f,0.f};
    v8f acc1 = acc0, acc2 = acc0, acc3 = acc0;
    union Frag { v16bf v; uint4 q[2]; };

    #pragma unroll
    for (int kk = 0; kk < C; kk += 32) {
        Frag fa;
        fa.q[0] = *(const uint4*)(arow + kk);
        fa.q[1] = *(const uint4*)(arow + kk + 16);
        Frag fb0, fb1, fb2, fb3;
        fb0.q[0] = *(const uint4*)(b0l + 0 * 16 * C + kk);
        fb0.q[1] = *(const uint4*)(b0l + 0 * 16 * C + kk + 16);
        fb1.q[0] = *(const uint4*)(b0l + 1 * 16 * C + kk);
        fb1.q[1] = *(const uint4*)(b0l + 1 * 16 * C + kk + 16);
        fb2.q[0] = *(const uint4*)(b0l + 2 * 16 * C + kk);
        fb2.q[1] = *(const uint4*)(b0l + 2 * 16 * C + kk + 16);
        fb3.q[0] = *(const uint4*)(b0l + 3 * 16 * C + kk);
        fb3.q[1] = *(const uint4*)(b0l + 3 * 16 * C + kk + 16);
        acc0 = __builtin_amdgcn_wmma_f32_16x16x32_bf16(false, fa.v, false, fb0.v, (short)0, acc0, false, false);
        acc1 = __builtin_amdgcn_wmma_f32_16x16x32_bf16(false, fa.v, false, fb1.v, (short)0, acc1, false, false);
        acc2 = __builtin_amdgcn_wmma_f32_16x16x32_bf16(false, fa.v, false, fb2.v, (short)0, acc2, false, false);
        acc3 = __builtin_amdgcn_wmma_f32_16x16x32_bf16(false, fa.v, false, fb3.v, (short)0, acc3, false, false);
    }

    // ---- epilogue: bias + ReLU + bf16 store
    const int nbase = blockIdx.y * 128 + nw * 64;
    v8f accs[4] = {acc0, acc1, acc2, acc3};
    #pragma unroll
    for (int j = 0; j < 4; ++j) {
        const int n = nbase + j * 16 + l16;
        const float bn = bias[n];
        #pragma unroll
        for (int r = 0; r < 8; ++r) {
            int m = m0 + r + hi * 8;   // D layout: VGPR r -> M=r (lanes 0-15), M=r+8 (16-31)
            float v = accs[j][r] + bn;
            v = v > 0.0f ? v : 0.0f;
            Y[(long)m * C + n] = (bf16_t)v;
        }
    }
}

// --------------------------------------------------- readout: mean/max over nodes
__global__ __launch_bounds__(256)
void k_readout(const bf16_t* __restrict__ x, float* __restrict__ hg) {
    const int b = blockIdx.x;
    const int c = threadIdx.x;                        // 256
    const bf16_t* xp = x + (long)b * N_NODES * C + c;
    float s = 0.0f, m = -3.4e38f;
    for (int n = 0; n < N_NODES; ++n) {
        float v = (float)xp[(long)n * C];
        s += v;
        m = fmaxf(m, v);
    }
    float mean = s * (1.0f / N_NODES);
    hg[b * HAZ + c]     += mean > 0.0f ? mean : 0.0f;
    hg[b * HAZ + C + c] += m    > 0.0f ? m    : 0.0f;
}

// ----------------------------------------------------------- fused MLP head
__global__ __launch_bounds__(512)
void k_mlp(const float* __restrict__ emb, const float* __restrict__ hg,
           const float* __restrict__ Wh1, const float* __restrict__ bh1,
           const float* __restrict__ Wh2, const float* __restrict__ bh2,
           const float* __restrict__ Wh3, float* __restrict__ out) {
    __shared__ float fus[C_EMB + HAZ];   // 640
    __shared__ float h1[HAZ];            // 512
    __shared__ float h2[HAZ / 2];        // 256
    __shared__ float red[HAZ / 2];
    const int b = blockIdx.x;
    const int t = threadIdx.x;

    for (int i = t; i < C_EMB + HAZ; i += 512)
        fus[i] = (i < C_EMB) ? emb[b * C_EMB + i] : hg[b * HAZ + (i - C_EMB)];
    __syncthreads();

    {   // h1 = relu(fus @ Wh1 + bh1), Wh1 is (640, 512)
        float acc = bh1[t];
        for (int k = 0; k < C_EMB + HAZ; ++k) acc += fus[k] * Wh1[k * HAZ + t];
        h1[t] = acc > 0.0f ? acc : 0.0f;
    }
    __syncthreads();

    if (t < HAZ / 2) {   // h2 = relu(h1 @ Wh2 + bh2), Wh2 is (512, 256)
        float acc = bh2[t];
        for (int k = 0; k < HAZ; ++k) acc += h1[k] * Wh2[k * (HAZ / 2) + t];
        h2[t] = acc > 0.0f ? acc : 0.0f;
    }
    __syncthreads();

    if (t < HAZ / 2) red[t] = h2[t] * Wh3[t];
    __syncthreads();
    for (int s = HAZ / 4; s > 0; s >>= 1) {
        if (t < s) red[t] += red[t + s];
        __syncthreads();
    }
    if (t == 0) out[b] = red[0];
}

// ------------------------------------------------------------------ launcher
extern "C" void kernel_launch(void* const* d_in, const int* in_sizes, int n_in,
                              void* d_out, int out_size, void* d_ws, size_t ws_size,
                              hipStream_t stream) {
    (void)in_sizes; (void)n_in; (void)out_size; (void)ws_size;
    const float* inp = (const float*)d_in[0];
    const int*   src = (const int*)d_in[1];
    const int*   dst = (const int*)d_in[2];
    const float* W0  = (const float*)d_in[3];
    const float* b0  = (const float*)d_in[4];
    const float* W1  = (const float*)d_in[5];
    const float* b1  = (const float*)d_in[6];
    const float* W2  = (const float*)d_in[7];
    const float* b2  = (const float*)d_in[8];
    const float* We  = (const float*)d_in[9];
    const float* Wh1 = (const float*)d_in[10];
    const float* bh1 = (const float*)d_in[11];
    const float* Wh2 = (const float*)d_in[12];
    const float* bh2 = (const float*)d_in[13];
    const float* Wh3 = (const float*)d_in[14];
    float* out = (float*)d_out;

    // ---- carve workspace (256B aligned slabs)
    uintptr_t p = (uintptr_t)d_ws;
    auto carve = [&](size_t bytes) -> void* {
        uintptr_t q = (p + 255) & ~(uintptr_t)255;
        p = q + bytes;
        return (void*)q;
    };
    int*    deg_src    = (int*)carve(N_NODES * sizeof(int));
    int*    deg_dst    = (int*)carve(N_NODES * sizeof(int));
    float*  dout_is    = (float*)carve(N_NODES * sizeof(float));
    float*  din_is     = (float*)carve(N_NODES * sizeof(float));
    int*    row_start  = (int*)carve((N_NODES + 1) * sizeof(int));
    int*    sorted_src = (int*)carve(NEDGE * sizeof(int));
    bf16_t* Wt1        = (bf16_t*)carve((size_t)C * C * sizeof(bf16_t));
    bf16_t* Wt2        = (bf16_t*)carve((size_t)C * C * sizeof(bf16_t));
    float*  emb        = (float*)carve((size_t)BATCH * C_EMB * sizeof(float));
    float*  hg         = (float*)carve((size_t)BATCH * HAZ * sizeof(float));
    bf16_t* xA         = (bf16_t*)carve((size_t)BATCH * N_NODES * C * sizeof(bf16_t));
    bf16_t* agg        = (bf16_t*)carve((size_t)BATCH * N_NODES * C * sizeof(bf16_t));

    const long M = (long)BATCH * N_NODES;             // 256000 GEMM rows

    // ---- graph structure + scales (deterministic)
    k_zero_i32<<<(N_NODES + 255) / 256, 256, 0, stream>>>(deg_src, N_NODES);
    k_zero_i32<<<(N_NODES + 255) / 256, 256, 0, stream>>>(deg_dst, N_NODES);
    k_zero_f32<<<(BATCH * HAZ + 255) / 256, 256, 0, stream>>>(hg, BATCH * HAZ);
    k_count<<<(NEDGE + 255) / 256, 256, 0, stream>>>(src, dst, deg_src, deg_dst);
    k_scan<<<1, 256, 0, stream>>>(deg_src, deg_dst, dout_is, din_is, row_start);
    k_bucket<<<(N_NODES + 255) / 256, 256, 0, stream>>>(src, dst, row_start, sorted_src);

    // ---- weight prep + embed
    k_prepW<<<C, C, 0, stream>>>(W1, Wt1);
    k_prepW<<<C, C, 0, stream>>>(W2, Wt2);
    k_embed<<<BATCH, C_EMB, 0, stream>>>(inp, We, emb);

    // ---- layer 0 (dim 1 -> 256) + readout
    k_layer0<<<dim3(N_NODES / 16, BATCH), 256, 0, stream>>>(
        inp, row_start, sorted_src, dout_is, din_is, W0, b0, xA);
    k_readout<<<BATCH, C, 0, stream>>>(xA, hg);

    // ---- layers 1 and 2: aggregate -> WMMA GEMM (in-place ping-pong) -> readout
    const dim3 gAgg(N_NODES / 8, BATCH);
    const dim3 gGemm((unsigned)(M / 64), C / 128);    // (4000, 2)

    k_aggregate<<<gAgg, 256, 0, stream>>>(xA, row_start, sorted_src, dout_is, din_is, agg);
    k_gemm_bf16<<<gGemm, 256, 0, stream>>>(agg, Wt1, b1, xA);
    k_readout<<<BATCH, C, 0, stream>>>(xA, hg);

    k_aggregate<<<gAgg, 256, 0, stream>>>(xA, row_start, sorted_src, dout_is, din_is, agg);
    k_gemm_bf16<<<gGemm, 256, 0, stream>>>(agg, Wt2, b2, xA);
    k_readout<<<BATCH, C, 0, stream>>>(xA, hg);

    // ---- fused MLP head
    k_mlp<<<BATCH, 512, 0, stream>>>(emb, hg, Wh1, bh1, Wh2, bh2, Wh3, out);
}
